// SepsisHANClassifier_70136815944151
// MI455X (gfx1250) — matrix operand order repaired and checked by reading, not compiled, not run on hardware
//
#include <hip/hip_runtime.h>
#include <hip/hip_bf16.h>
#include <math.h>

typedef _Float16 half_t;
typedef __attribute__((ext_vector_type(16))) _Float16 v16h;
typedef __attribute__((ext_vector_type(8)))  _Float16 v8h;
typedef __attribute__((ext_vector_type(8)))  float    v8f;

#define HEADS 8
#define NEG_SLOPE 0.2f
#define TB 256

static __device__ __forceinline__ void atomicMaxF(float* addr, float val) {
  unsigned int* ua = (unsigned int*)addr;
  unsigned int old = __float_as_uint(*addr);
  while (__uint_as_float(old) < val) {
    unsigned int assumed = old;
    old = atomicCAS(ua, assumed, __float_as_uint(val));
    if (old == assumed) break;
  }
}

__global__ void k_fill(float* p, float v, int n) {
  int i = blockIdx.x * blockDim.x + threadIdx.x;
  if (i < n) p[i] = v;
}

__global__ void k_cvt_f16(const float* __restrict__ src, half_t* __restrict__ dst, int n) {
  int i = blockIdx.x * blockDim.x + threadIdx.x;
  if (i < n) dst[i] = (half_t)src[i];
}

// Convert W (f32, KxN row-major) into f16 WMMA B-fragment order:
// fragment f = (kb, tn) holds 512 halves; within it lane L (0..31) owns 16
// contiguous halves: B[kb*32 + (i<8 ? (L>=16?8:0) + i : 16 + (L>=16?8:0) + i-8)][tn*16 + (L&15)], i=0..15.
__global__ void k_cvt_w_frag(const float* __restrict__ W, half_t* __restrict__ out,
                             int K, int N) {
  int i = blockIdx.x * blockDim.x + threadIdx.x;
  if (i >= K * N) return;
  int ntn  = N >> 4;
  int frag = i >> 9;
  int rem  = i & 511;
  int lane = rem >> 4;
  int ii   = rem & 15;
  int kb   = frag / ntn;
  int tn   = frag % ntn;
  int k = kb * 32 + (ii < 8 ? (lane >> 4) * 8 + ii : 16 + (lane >> 4) * 8 + (ii - 8));
  int n = tn * 16 + (lane & 15);
  out[i] = (half_t)W[k * N + n];
}

// C[M,N] = A[M,K] @ W[K,N] + bias[N]  (optional relu).
// A: f16 row-major; WP: f16 pre-swizzled B-fragments (see k_cvt_w_frag).
// One wave per 16x16 tile; K multiple of 32; M,N multiples of 16.
__global__ void k_wmma_gemm(const half_t* __restrict__ A, const half_t* __restrict__ WP,
                            const float* __restrict__ bias, float* __restrict__ C,
                            int M, int N, int K, int do_relu) {
  int lane = threadIdx.x & 31;
  int tile = blockIdx.x * (blockDim.x >> 5) + (threadIdx.x >> 5);
  int ntn  = N >> 4;
  int tm = tile / ntn, tn = tile % ntn;
  if (tm * 16 >= M) return;
  int m0 = tm * 16, n0 = tn * 16;
  int hs  = lane >> 4;        // lane half select
  int l15 = lane & 15;
  int arow = m0 + l15;
  int bcol = n0 + l15;
  v8f acc = {};
  for (int kb = 0; kb < K; kb += 32) {
    // A 16x32 f16 layout: lanes<16: K = kb+0..7 and kb+16..23 ; lanes>=16: +8
    const half_t* ap = A + (size_t)arow * K + kb + hs * 8;
    v8h alo = *(const v8h*)(ap);
    v8h ahi = *(const v8h*)(ap + 16);
    v16h a;
#pragma unroll
    for (int i = 0; i < 8; i++) { a[i] = alo[i]; a[8 + i] = ahi[i]; }
    // B fragment: contiguous 32B per lane in packed layout
    const half_t* bp = WP + ((size_t)((kb >> 5) * ntn + tn) << 9) + (lane << 4);
    v16h b = *(const v16h*)bp;
    acc = __builtin_amdgcn_wmma_f32_16x16x32_f16(false, a, false, b, (short)0, acc, false, false);
  }
  // D: lanes<16: M=m0+r, N=n0+lane ; lanes>=16: M=m0+8+r
  int mbase = m0 + hs * 8;
  float bv = bias[bcol];
#pragma unroll
  for (int r = 0; r < 8; r++) {
    float v = acc[r] + bv;
    if (do_relu) v = fmaxf(v, 0.0f);
    C[(size_t)(mbase + r) * N + bcol] = v;
  }
}

__global__ void k_relu_copy(const float* __restrict__ in, float* __restrict__ out, int n) {
  int i = blockIdx.x * blockDim.x + threadIdx.x;
  if (i < n) out[i] = fmaxf(in[i], 0.0f);
}

// per-node attention coefficient: out[n*8+h] = sum_j xp[n*C + h*d + j] * att[h*d + j]
__global__ void k_node_att(const float* __restrict__ xp, const float* __restrict__ att,
                           float* __restrict__ out, int n_nodes, int C, int d) {
  int i = blockIdx.x * blockDim.x + threadIdx.x;
  if (i >= n_nodes * HEADS) return;
  int n = i >> 3, h = i & 7;
  const float* xr = xp + (size_t)n * C + h * d;
  const float* ar = att + h * d;
  float s = 0.0f;
  for (int j = 0; j < d; j++) s += xr[j] * ar[j];
  out[i] = s;
}

__global__ void k_edge_max(const int* __restrict__ src, const int* __restrict__ dst,
                           const float* __restrict__ asrc, const float* __restrict__ adst,
                           float* __restrict__ smax, int E) {
  int i = blockIdx.x * blockDim.x + threadIdx.x;
  if (i >= E * HEADS) return;
  int e = i >> 3, h = i & 7;
  int s = src[e], t = dst[e];
  float al = asrc[s * HEADS + h] + adst[t * HEADS + h];
  al = al > 0.0f ? al : NEG_SLOPE * al;
  atomicMaxF(&smax[t * HEADS + h], al);
}

// accumulate out[dst] += e * xp[src], ssum[dst] += e   (normalize later)
__global__ void k_edge_msg(const int* __restrict__ src, const int* __restrict__ dst,
                           const float* __restrict__ asrc, const float* __restrict__ adst,
                           const float* __restrict__ smax, const float* __restrict__ xp,
                           float* __restrict__ outbuf, float* __restrict__ ssum,
                           int E, int C, int d) {
  int i = blockIdx.x * blockDim.x + threadIdx.x;
  if (i >= E * HEADS) return;
  int e = i >> 3, h = i & 7;
  int s = src[e], t = dst[e];
  float al = asrc[s * HEADS + h] + adst[t * HEADS + h];
  al = al > 0.0f ? al : NEG_SLOPE * al;
  float ex = expf(al - smax[t * HEADS + h]);
  atomicAdd(&ssum[t * HEADS + h], ex);
  const float* xr = xp + (size_t)s * C + h * d;
  float* orow = outbuf + (size_t)t * C + h * d;
  for (int j = 0; j < d; j++) atomicAdd(&orow[j], ex * xr[j]);
}

__global__ void k_finalize(float* __restrict__ outbuf, const float* __restrict__ ssum,
                           int n_nodes, int C, int d) {
  int i = blockIdx.x * blockDim.x + threadIdx.x;
  if (i >= n_nodes * C) return;
  int n = i / C, c = i % C, h = c / d;
  float v = outbuf[i] / (ssum[n * HEADS + h] + 1e-16f);
  outbuf[i] = fmaxf(v, 0.0f);
}

// meanbuf[c] += (1/N) * sum over rows of tanh(dot(stk[n], kw[:,c]) + kb[c])
__global__ void k_sem_mean(const float* __restrict__ stk, const float* __restrict__ kw,
                           const float* __restrict__ kb, float* __restrict__ meanbuf,
                           int N, int C, int rows_per_block) {
  int c = threadIdx.x;
  int n0 = blockIdx.x * rows_per_block;
  float acc = 0.0f;
  for (int r = 0; r < rows_per_block; r++) {
    int n = n0 + r;
    if (n >= N) break;
    const float* row = stk + (size_t)n * C;
    float s = kb[c];
    for (int j = 0; j < C; j++) s += row[j] * kw[j * C + c];
    acc += tanhf(s);
  }
  atomicAdd(&meanbuf[c], acc / (float)N);
}

__global__ void k_sem_attn(const float* __restrict__ meanbuf, const float* __restrict__ q,
                           float* __restrict__ attn, int K, int C) {
  if (threadIdx.x == 0) {
    float sc[4];
    float mx = -1e30f;
    for (int k = 0; k < K; k++) {
      float s = 0.0f;
      for (int c = 0; c < C; c++) s += q[c] * meanbuf[k * C + c];
      sc[k] = s;
      if (s > mx) mx = s;
    }
    float tot = 0.0f;
    for (int k = 0; k < K; k++) { sc[k] = expf(sc[k] - mx); tot += sc[k]; }
    for (int k = 0; k < K; k++) attn[k] = sc[k] / tot;
  }
}

__global__ void k_sem_combine(const float* __restrict__ s0, const float* __restrict__ s1,
                              const float* __restrict__ s2, const float* __restrict__ attn,
                              float* __restrict__ out, int N, int C, int K) {
  int i = blockIdx.x * blockDim.x + threadIdx.x;
  if (i >= N * C) return;
  float v = attn[0] * s0[i] + attn[1] * s1[i];
  if (K > 2) v += attn[2] * s2[i];
  out[i] = v;
}

__global__ void k_cls_logits(const float* __restrict__ h1, const float* __restrict__ w2,
                             const float* __restrict__ b2, float* __restrict__ logits, int N) {
  int i = blockIdx.x * blockDim.x + threadIdx.x;
  if (i >= N * 2) return;
  int n = i >> 1, c = i & 1;
  float s = b2[c];
  const float* hr = h1 + (size_t)n * 32;
  for (int j = 0; j < 32; j++) s += hr[j] * w2[j * 2 + c];
  logits[i] = s;
}

// ----------------------------- host side -----------------------------------

static inline int cdiv(long long a, long long b) { return (int)((a + b - 1) / b); }

static void run_gemm(const float* A, const float* W, const float* bias, float* C,
                     int M, int N, int K, int relu, half_t* AF16, half_t* WF16,
                     hipStream_t st) {
  k_cvt_f16<<<cdiv((long long)M * K, TB), TB, 0, st>>>(A, AF16, M * K);
  k_cvt_w_frag<<<cdiv((long long)K * N, TB), TB, 0, st>>>(W, WF16, K, N);
  int tiles = (M / 16) * (N / 16);
  k_wmma_gemm<<<cdiv(tiles, 4), 128, 0, st>>>(AF16, WF16, bias, C, M, N, K, relu);
}

static void run_edge(const int* src, const int* dst, int E,
                     const float* xp_src, const float* xp_dst,
                     int Nsrc, int Ndst, int C, int d,
                     const float* att_s, const float* att_d, float* slot,
                     float* ASRC, float* ADST, float* SMAX, float* SSUM, hipStream_t st) {
  k_node_att<<<cdiv((long long)Nsrc * HEADS, TB), TB, 0, st>>>(xp_src, att_s, ASRC, Nsrc, C, d);
  k_node_att<<<cdiv((long long)Ndst * HEADS, TB), TB, 0, st>>>(xp_dst, att_d, ADST, Ndst, C, d);
  k_fill<<<cdiv((long long)Ndst * HEADS, TB), TB, 0, st>>>(SMAX, -INFINITY, Ndst * HEADS);
  k_fill<<<cdiv((long long)Ndst * HEADS, TB), TB, 0, st>>>(SSUM, 0.0f, Ndst * HEADS);
  k_fill<<<cdiv((long long)Ndst * C, TB), TB, 0, st>>>(slot, 0.0f, Ndst * C);
  k_edge_max<<<cdiv((long long)E * HEADS, TB), TB, 0, st>>>(src, dst, ASRC, ADST, SMAX, E);
  k_edge_msg<<<cdiv((long long)E * HEADS, TB), TB, 0, st>>>(src, dst, ASRC, ADST, SMAX, xp_src,
                                                            slot, SSUM, E, C, d);
  k_finalize<<<cdiv((long long)Ndst * C, TB), TB, 0, st>>>(slot, SSUM, Ndst, C, d);
}

static void run_sem(float* s0, float* s1, float* s2, int K, int N, int C,
                    const float* kw, const float* kb, const float* q,
                    float* SEM, float* out, hipStream_t st) {
  k_fill<<<1, 256, 0, st>>>(SEM, 0.0f, 256);
  const int RPB = 64;
  float* slots[3] = {s0, s1, s2};
  for (int k = 0; k < K; k++)
    k_sem_mean<<<cdiv(N, RPB), C, 0, st>>>(slots[k], kw, kb, SEM + k * C, N, C, RPB);
  k_sem_attn<<<1, 32, 0, st>>>(SEM, q, SEM + 192, K, C);
  k_sem_combine<<<cdiv((long long)N * C, TB), TB, 0, st>>>(s0, s1, K > 2 ? s2 : s1,
                                                           SEM + 192, out, N, C, K);
}

static void run_han_layer(const float* hS, const float* hG, const float* hP,
                          int CinS, int CinG, int CinP, int Cout,
                          void* const* d_in, int L,
                          const int* eiExp, int E1, const int* eiEnc, int E2,
                          float* XP_S, float* XP_G, float* XP_P,
                          float* S0, float* S1, float* G0, float* G1, float* G2,
                          float* P0, float* P1,
                          float* ASRC, float* ADST, float* SMAX, float* SSUM, float* SEM,
                          half_t* AF16, half_t* WF16,
                          float* outS, float* outG, float* outP, hipStream_t st) {
  const int NS = 50000, NG = 20000, NP = 20000;
  int d = Cout / HEADS;
  auto P = [&](int i) { return (const float*)d_in[i]; };
  // leaf order within layer: att_dst[7]=L.., att_src[7]=L+7..,
  // k_b=L+14, k_w=L+15, proj G{b,w}=L+16,17, P{b,w}=L+18,19, S{b,w}=L+20,21, q=L+22
  run_gemm(hS, P(L + 21), P(L + 20), XP_S, NS, Cout, CinS, 0, AF16, WF16, st);
  run_gemm(hG, P(L + 17), P(L + 16), XP_G, NG, Cout, CinG, 0, AF16, WF16, st);
  run_gemm(hP, P(L + 19), P(L + 18), XP_P, NP, Cout, CinP, 0, AF16, WF16, st);
  // sorted ekey positions: encodes=0, expresses_rev=1, selfG=2, encodes_rev=3,
  //                        selfP=4, expresses=5, selfS=6
  // expresses: Sample->Gene, slot G0
  run_edge(eiExp, eiExp + E1, E1, XP_S, XP_G, NS, NG, Cout, d,
           P(L + 7 + 5), P(L + 5), G0, ASRC, ADST, SMAX, SSUM, st);
  // encodes: Gene->Protein, slot P0
  run_edge(eiEnc, eiEnc + E2, E2, XP_G, XP_P, NG, NP, Cout, d,
           P(L + 7 + 0), P(L + 0), P0, ASRC, ADST, SMAX, SSUM, st);
  // expresses_rev: Gene->Sample, slot S0
  run_edge(eiExp + E1, eiExp, E1, XP_G, XP_S, NG, NS, Cout, d,
           P(L + 7 + 1), P(L + 1), S0, ASRC, ADST, SMAX, SSUM, st);
  // encodes_rev: Protein->Gene, slot G1
  run_edge(eiEnc + E2, eiEnc, E2, XP_P, XP_G, NP, NG, Cout, d,
           P(L + 7 + 3), P(L + 3), G1, ASRC, ADST, SMAX, SSUM, st);
  // self loops: softmax over single edge == 1 -> relu(xp)
  k_relu_copy<<<cdiv((long long)NS * Cout, TB), TB, 0, st>>>(XP_S, S1, NS * Cout);
  k_relu_copy<<<cdiv((long long)NG * Cout, TB), TB, 0, st>>>(XP_G, G2, NG * Cout);
  k_relu_copy<<<cdiv((long long)NP * Cout, TB), TB, 0, st>>>(XP_P, P1, NP * Cout);
  // semantic attention (stack order per reference append order)
  run_sem(S0, S1, nullptr, 2, NS, Cout, P(L + 15), P(L + 14), P(L + 22), SEM, outS, st);
  run_sem(G0, G1, G2, 3, NG, Cout, P(L + 15), P(L + 14), P(L + 22), SEM, outG, st);
  run_sem(P0, P1, nullptr, 2, NP, Cout, P(L + 15), P(L + 14), P(L + 22), SEM, outP, st);
}

extern "C" void kernel_launch(void* const* d_in, const int* in_sizes, int n_in,
                              void* d_out, int out_size, void* d_ws, size_t ws_size,
                              hipStream_t stream) {
  const int NS = 50000, NG = 20000, NP = 20000;
  const float* xS = (const float*)d_in[0];
  const float* xG = (const float*)d_in[1];
  const float* xP = (const float*)d_in[2];
  const int* eiExp = (const int*)d_in[3];
  const int* eiEnc = (const int*)d_in[4];
  int E1 = in_sizes[3] / 2;
  int E2 = in_sizes[4] / 2;
  // params leaves (sorted-dict flatten): 5 cls.b1, 6 cls.b2, 7 cls.w1, 8 cls.w2,
  // layer0 base 9, layer1 base 32
  auto P = [&](int i) { return (const float*)d_in[i]; };

  float* fw = (float*)d_ws;
  size_t off = 0;
  auto alloc = [&](size_t n) { float* p = fw + off; off += n; return p; };
  float* XP_S = alloc((size_t)NS * 64);
  float* XP_G = alloc((size_t)NG * 64);
  float* XP_P = alloc((size_t)NP * 64);
  float* HN_S = alloc((size_t)NS * 64);
  float* HN_G = alloc((size_t)NG * 64);
  float* HN_P = alloc((size_t)NP * 64);
  float* S0 = alloc((size_t)NS * 64);
  float* S1 = alloc((size_t)NS * 64);
  float* G0 = alloc((size_t)NG * 64);
  float* G1 = alloc((size_t)NG * 64);
  float* G2 = alloc((size_t)NG * 64);
  float* P0 = alloc((size_t)NP * 64);
  float* P1 = alloc((size_t)NP * 64);
  float* ASRC = alloc((size_t)NS * 8);
  float* ADST = alloc((size_t)NS * 8);
  float* SMAX = alloc((size_t)NS * 8);
  float* SSUM = alloc((size_t)NS * 8);
  float* SEM = alloc(256);
  float* H1B = alloc((size_t)NS * 32);
  half_t* AF16 = (half_t*)alloc((size_t)NS * 32);  // NS*64 halves
  half_t* WF16 = (half_t*)alloc(2048);             // 4096 halves

  float* out = (float*)d_out;
  float* emb = out + (size_t)NS * 2;  // layer1 Sample embedding, 50000x32

  // layer 0: in dims {S:64, G:32, P:32} -> 64
  run_han_layer(xS, xG, xP, 64, 32, 32, 64, d_in, 9, eiExp, E1, eiEnc, E2,
                XP_S, XP_G, XP_P, S0, S1, G0, G1, G2, P0, P1,
                ASRC, ADST, SMAX, SSUM, SEM, AF16, WF16,
                HN_S, HN_G, HN_P, stream);
  // layer 1: in dims 64 -> 32 ; Sample output written directly to emb region
  run_han_layer(HN_S, HN_G, HN_P, 64, 64, 64, 32, d_in, 32, eiExp, E1, eiEnc, E2,
                XP_S, XP_G, XP_P, S0, S1, G0, G1, G2, P0, P1,
                ASRC, ADST, SMAX, SSUM, SEM, AF16, WF16,
                emb, HN_G, HN_P, stream);
  // classifier: h1 = relu(emb @ w1 + b1) ; logits = h1 @ w2 + b2
  run_gemm(emb, P(7), P(5), H1B, NS, 32, 32, 1, AF16, WF16, stream);
  k_cls_logits<<<cdiv((long long)NS * 2, TB), TB, 0, stream>>>(H1B, P(8), P(6), out, NS);
}
